// ClusteringLoss_75505525064683
// MI455X (gfx1250) — matrix operand order repaired and checked by reading, not compiled, not run on hardware
//
#include <hip/hip_runtime.h>
#include <hip/hip_bf16.h>

// ---------------------------------------------------------------------------
// Problem dims (fixed by the reference): B=16, S=2048, D=512, K=1024
// M = B*S = 32768 rows of features; GEMM: [M,D] x [D,K] -> [M,K]
// ---------------------------------------------------------------------------
#define DIM_M 32768
#define DIM_D 512
#define DIM_K 1024

typedef __attribute__((ext_vector_type(16))) __bf16          v16bf;
typedef __attribute__((ext_vector_type(8)))  float           v8f;
typedef __attribute__((ext_vector_type(8)))  unsigned short  u16x8;
typedef __attribute__((ext_vector_type(4)))  unsigned short  u16x4;

union Frag16 {            // bit-assemble a WMMA 16-bit operand from LDS halves
    v16bf  bf;
    u16x8  u8[2];
};

// round-to-nearest-even f32 -> bf16 (bit pattern)
static __device__ __forceinline__ unsigned short f32_to_bf16_rne(float f) {
    unsigned int u = __float_as_uint(f);
    unsigned int r = u + 0x7FFFu + ((u >> 16) & 1u);
    return (unsigned short)(r >> 16);
}
static __device__ __forceinline__ float bf16_hi_to_f32(unsigned short h) {
    return __uint_as_float(((unsigned int)h) << 16);
}

// ---------------------------------------------------------------------------
// rowsq: one wave per row, exact f32 sum of squares (for x^2 and c^2 terms)
// ---------------------------------------------------------------------------
__global__ void rowsq_kernel(const float* __restrict__ src,
                             float* __restrict__ dst, int nrows) {
    const int lane = threadIdx.x & 31;
    const int wave = threadIdx.x >> 5;
    const int row  = blockIdx.x * 8 + wave;
    if (row >= nrows) return;
    const float4* p = (const float4*)(src + (size_t)row * DIM_D);
    float s = 0.0f;
#pragma unroll
    for (int i = lane; i < DIM_D / 4; i += 32) {
        float4 v = p[i];
        s += v.x * v.x + v.y * v.y + v.z * v.z + v.w * v.w;
    }
#pragma unroll
    for (int off = 16; off > 0; off >>= 1) s += __shfl_xor(s, off, 32);
    if (lane == 0) dst[row] = s;
}

// ---------------------------------------------------------------------------
// Split-BF16 WMMA GEMM + distance epilogue.
//   block tile: 128(M) x 128(N), K-step 32
//   8 waves: waveM in 0..3 (32 rows each), waveN in 0..1 (64 cols each)
//   per wave: 2x4 = 8 WMMA accumulator tiles, 3 WMMAs per tile per K-step
// ---------------------------------------------------------------------------
#define BM 128
#define BN 128
#define BK 32
#define LDA 40    // padded bf16 stride for A tiles (80B rows -> conflict-free)
#define LDB 136   // padded bf16 stride for B tiles (272B rows -> 2-way max)

__global__ __launch_bounds__(256)
void cluster_dist_wmma_kernel(const float* __restrict__ X,
                              const float* __restrict__ C,
                              const float* __restrict__ xsq,
                              const float* __restrict__ csq,
                              float* __restrict__ out) {
    __shared__ unsigned short sAh[BM * LDA];
    __shared__ unsigned short sAl[BM * LDA];
    __shared__ unsigned short sBh[BK * LDB];
    __shared__ unsigned short sBl[BK * LDB];

    const int tid   = threadIdx.x;
    const int lane  = tid & 31;
    const int wave  = tid >> 5;
    const int waveM = wave >> 1;   // 0..3
    const int waveN = wave & 1;    // 0..1
    const int blockM = blockIdx.x * BM;
    const int blockN = blockIdx.y * BN;

    v8f acc[2][4] = {};

    const int mloc = lane & 15;
    const int kb   = (lane >> 4) << 3;     // 0 or 8 (A-fragment K swizzle)
    const int krow = lane & 15;            // B fragment: lane == N column, elements over K

    for (int k0 = 0; k0 < DIM_D; k0 += BK) {
        // ---- stage A tile: 128x32 f32 -> hi/lo bf16 planes in LDS --------
#pragma unroll
        for (int i = 0; i < 4; ++i) {
            const int f   = tid + 256 * i;
            const int row = f >> 3;        // 8 float4 per 32-float row
            const int c4  = f & 7;
            const float4 v = *(const float4*)(X + (size_t)(blockM + row) * DIM_D + k0 + c4 * 4);
            u16x4 hi, lo;
            float e;
            hi.x = f32_to_bf16_rne(v.x); e = v.x - bf16_hi_to_f32(hi.x); lo.x = f32_to_bf16_rne(e);
            hi.y = f32_to_bf16_rne(v.y); e = v.y - bf16_hi_to_f32(hi.y); lo.y = f32_to_bf16_rne(e);
            hi.z = f32_to_bf16_rne(v.z); e = v.z - bf16_hi_to_f32(hi.z); lo.z = f32_to_bf16_rne(e);
            hi.w = f32_to_bf16_rne(v.w); e = v.w - bf16_hi_to_f32(hi.w); lo.w = f32_to_bf16_rne(e);
            *(u16x4*)&sAh[row * LDA + c4 * 4] = hi;
            *(u16x4*)&sAl[row * LDA + c4 * 4] = lo;
        }
        // ---- stage B tile: 128 clusters x 32 dims, transposed to [K][N] --
#pragma unroll
        for (int i = 0; i < 4; ++i) {
            const int f  = tid + 256 * i;
            const int n  = f >> 3;
            const int k4 = f & 7;
            const float4 v = *(const float4*)(C + (size_t)(blockN + n) * DIM_D + k0 + k4 * 4);
            const float vv[4] = {v.x, v.y, v.z, v.w};
#pragma unroll
            for (int j = 0; j < 4; ++j) {
                const unsigned short h = f32_to_bf16_rne(vv[j]);
                const unsigned short l = f32_to_bf16_rne(vv[j] - bf16_hi_to_f32(h));
                sBh[(k4 * 4 + j) * LDB + n] = h;
                sBl[(k4 * 4 + j) * LDB + n] = l;
            }
        }
        // prefetch next K-tile while waves sync
        if (k0 + BK < DIM_D) {
            __builtin_prefetch(X + (size_t)(blockM + (tid >> 3)) * DIM_D + k0 + BK, 0, 0);
            __builtin_prefetch(C + (size_t)(blockN + (tid >> 3)) * DIM_D + k0 + BK, 0, 0);
        }
        __syncthreads();

        // ---- build fragments (ISA 7.12.2 layouts) ------------------------
        Frag16 aH[2], aL[2];
#pragma unroll
        for (int mt = 0; mt < 2; ++mt) {
            const int r = (waveM * 32 + mt * 16 + mloc) * LDA;
            aH[mt].u8[0] = *(const u16x8*)&sAh[r + kb];
            aH[mt].u8[1] = *(const u16x8*)&sAh[r + 16 + kb];
            aL[mt].u8[0] = *(const u16x8*)&sAl[r + kb];
            aL[mt].u8[1] = *(const u16x8*)&sAl[r + 16 + kb];
        }
        Frag16 bH[4], bL[4];
#pragma unroll
        for (int nt = 0; nt < 4; ++nt) {
            const int col = waveN * 64 + nt * 16;
#pragma unroll
            for (int s = 0; s < 8; ++s) {
                bH[nt].u8[0][s] = sBh[(kb + s) * LDB + col + krow];
                bH[nt].u8[1][s] = sBh[(16 + kb + s) * LDB + col + krow];
                bL[nt].u8[0][s] = sBl[(kb + s) * LDB + col + krow];
                bL[nt].u8[1][s] = sBl[(16 + kb + s) * LDB + col + krow];
            }
        }

        // ---- 3-term split-bf16 MACC: hi*hi + hi*lo + lo*hi ---------------
#pragma unroll
        for (int mt = 0; mt < 2; ++mt) {
#pragma unroll
            for (int nt = 0; nt < 4; ++nt) {
                acc[mt][nt] = __builtin_amdgcn_wmma_f32_16x16x32_bf16(
                    false, aH[mt].bf, false, bH[nt].bf, (short)0, acc[mt][nt], false, false);
                acc[mt][nt] = __builtin_amdgcn_wmma_f32_16x16x32_bf16(
                    false, aH[mt].bf, false, bL[nt].bf, (short)0, acc[mt][nt], false, false);
                acc[mt][nt] = __builtin_amdgcn_wmma_f32_16x16x32_bf16(
                    false, aL[mt].bf, false, bH[nt].bf, (short)0, acc[mt][nt], false, false);
            }
        }
        __syncthreads();
    }

    // ---- epilogue: dist = x^2 + c^2 - 2 * (x . c) ------------------------
#pragma unroll
    for (int mt = 0; mt < 2; ++mt) {
        const int m0 = blockM + waveM * 32 + mt * 16 + (lane >> 4) * 8;
#pragma unroll
        for (int nt = 0; nt < 4; ++nt) {
            const int n   = blockN + waveN * 64 + nt * 16 + (lane & 15);
            const float c2n = csq[n];
            const v8f a = acc[mt][nt];
#pragma unroll
            for (int e = 0; e < 8; ++e) {
                const int m = m0 + e;
                out[(size_t)m * DIM_K + n] = xsq[m] + c2n - 2.0f * a[e];
            }
        }
    }
}

// ---------------------------------------------------------------------------
extern "C" void kernel_launch(void* const* d_in, const int* in_sizes, int n_in,
                              void* d_out, int out_size, void* d_ws, size_t ws_size,
                              hipStream_t stream) {
    const float* features = (const float*)d_in[0];   // [16,2048,512] f32
    const float* Ck       = (const float*)d_in[1];   // [1,1024,512] f32
    float* out = (float*)d_out;                      // [16,2048,1024] f32

    float* xsq = (float*)d_ws;                       // DIM_M floats
    float* csq = xsq + DIM_M;                        // DIM_K floats
    (void)in_sizes; (void)n_in; (void)out_size; (void)ws_size;

    // exact f32 row norms
    rowsq_kernel<<<DIM_M / 8, 256, 0, stream>>>(features, xsq, DIM_M);
    rowsq_kernel<<<DIM_K / 8, 256, 0, stream>>>(Ck, csq, DIM_K);

    // WMMA GEMM + fused distance epilogue
    dim3 grid(DIM_M / BM, DIM_K / BN);
    cluster_dist_wmma_kernel<<<grid, dim3(256), 0, stream>>>(features, Ck, xsq, csq, out);
}
